// PredictionModel_81570018885820
// MI455X (gfx1250) — matrix operand order repaired and checked by reading, not compiled, not run on hardware
//
#include <hip/hip_runtime.h>
#include <hip/hip_bf16.h>

#define BB 128
#define QQ 512
#define CC 32
#define EE 128
#define HHD 8

typedef unsigned short bfu;
typedef float v8f  __attribute__((ext_vector_type(8)));
typedef __bf16 v16bf __attribute__((ext_vector_type(16)));
typedef bfu u16x8  __attribute__((ext_vector_type(8)));
typedef bfu u16x16 __attribute__((ext_vector_type(16)));

__device__ __forceinline__ bfu f2bf(float f) {
    unsigned u = __builtin_bit_cast(unsigned, f);
    u += 0x7FFFu + ((u >> 16) & 1u);
    return (bfu)(u >> 16);
}
__device__ __forceinline__ float bf2f(bfu h) {
    return __builtin_bit_cast(float, ((unsigned)h) << 16);
}

// ---------------------------------------------------------------------------
// Generic bf16 WMMA GEMM: D[M,N] = act(A[M,K] @ W[N,K]^T + bias)
// A: bf16 row-major (lda), W: bf16 [N,K] row-major, bias: f32 or null.
// Block = 256 threads = 8 waves; wave w -> rows [bx*128 + 16w, +16);
// blockIdx.y -> cols [by*16, +16). M % 128 == 0, N % 16 == 0, K % 32 == 0.
// ---------------------------------------------------------------------------
__global__ __launch_bounds__(256)
void gemm_bf16_wmma(const bfu* __restrict__ A, int lda,
                    const bfu* __restrict__ W,
                    const float* __restrict__ bias,
                    int K, int relu,
                    bfu* __restrict__ Dbf, int ldd,
                    float* __restrict__ Df, int ldf) {
    const int lane  = threadIdx.x & 31;
    const int wave  = threadIdx.x >> 5;
    const int mbase = blockIdx.x * 128 + wave * 16;
    const int nbase = blockIdx.y * 16;
    const int lhalf = lane >> 4;   // 0/1: which K-half this lane supplies
    const int lmod  = lane & 15;

    v8f acc = {};
    // A 16x32 bf16 layout: lane (L&15)=row, (L>>4) selects K chunks {h*8..h*8+7, h*8+16..h*8+23}
    const bfu* arow = A + (size_t)(mbase + lmod) * lda + lhalf * 8;
    // B 32x16 bf16 layout: lane (L&15)=col n, (L>>4) selects K chunks {h*8..h*8+7, h*8+16..h*8+23}; B[k][n] = W[n][k]
    const bfu* wrow = W + (size_t)(nbase + lmod) * K + lhalf * 8;

    for (int kk = 0; kk < K; kk += 32) {
        union { u16x8 h[2]; u16x16 v; } au;
        au.h[0] = *reinterpret_cast<const u16x8*>(arow + kk);
        au.h[1] = *reinterpret_cast<const u16x8*>(arow + kk + 16);
        union { u16x8 h[2]; u16x16 v; } buu;
        buu.h[0] = *reinterpret_cast<const u16x8*>(wrow + kk);
        buu.h[1] = *reinterpret_cast<const u16x8*>(wrow + kk + 16);
        u16x16 bu = buu.v;
        if (kk + 32 < K) {  // gfx1250 global_prefetch_b8 path
            __builtin_prefetch(arow + kk + 32, 0, 1);
            __builtin_prefetch(wrow + kk + 32, 0, 1);
        }
        v16bf av = __builtin_bit_cast(v16bf, au.v);
        v16bf bv = __builtin_bit_cast(v16bf, bu);
        acc = __builtin_amdgcn_wmma_f32_16x16x32_bf16(
                  false, av, false, bv, (short)0, acc, false, false);
    }
    // C/D layout: VGPR r, lane L -> row = r + 8*(L>>4), col = L&15
    const int n  = nbase + lmod;
    const float bs = bias ? bias[n] : 0.0f;
#pragma unroll
    for (int r = 0; r < 8; ++r) {
        const int m = mbase + r + 8 * lhalf;
        float v = acc[r] + bs;
        if (relu) v = v > 0.f ? v : 0.f;
        if (Dbf) Dbf[(size_t)m * ldd + n] = f2bf(v);
        if (Df)  Df [(size_t)m * ldf + n] = v;
    }
}

// ---------------------------------------------------------------------------
__global__ void cvt_f32_bf16(const float* __restrict__ s, bfu* __restrict__ d, int n) {
    for (int i = blockIdx.x * blockDim.x + threadIdx.x; i < n;
         i += gridDim.x * blockDim.x)
        d[i] = f2bf(s[i]);
}

// ---------------------------------------------------------------------------
// Per-batch prep: has_prev / dbl / pq0 / pq1, and the qm query mask (bf16).
// ---------------------------------------------------------------------------
__global__ void batch_prep(const int* __restrict__ qubits,
                           const int* __restrict__ prev,
                           int* __restrict__ meta, bfu* __restrict__ qm) {
    const int b = blockIdx.x, t = threadIdx.x;  // blockDim 128
    __shared__ int flag;
    if (t == 0) flag = 0;
    __syncthreads();
    for (int q = t; q < QQ; q += 128)
        if (prev[b * QQ + q] != CC) flag = 1;
    __syncthreads();
    const int hp = flag;
    const int q0 = qubits[2 * b], q1 = qubits[2 * b + 1];
    if (t == 0) {
        int pq0 = prev[b * QQ + q0];
        pq0 = pq0 < 0 ? 0 : (pq0 > CC - 1 ? CC - 1 : pq0);
        int q1c = q1 < 0 ? 0 : q1;
        int pq1 = prev[b * QQ + q1c];
        pq1 = pq1 < 0 ? 0 : (pq1 > CC - 1 ? CC - 1 : pq1);
        meta[4 * b + 0] = hp;
        meta[4 * b + 1] = (q1 != -1) && hp;
        meta[4 * b + 2] = pq0;
        meta[4 * b + 3] = pq1;
    }
    for (int q = t; q < QQ; q += 128) {
        float v = (q == q0 || (q1 != -1 && q == q1)) ? 1.f : 0.f;
        qm[(size_t)b * QQ + q] = f2bf(v);
    }
}

// ---------------------------------------------------------------------------
// number-embedding MLP (1 -> 64 -> 128) for 2*B*C scalars.
// num_emb stored as [B, 2C, E] f32 (== [B, C, 2E] flat, matching reference).
// ---------------------------------------------------------------------------
__global__ void num_emb_kernel(const float* __restrict__ conn,
                               const float* __restrict__ caps,
                               const int* __restrict__ meta,
                               const float* __restrict__ W1, const float* __restrict__ b1,
                               const float* __restrict__ W2, const float* __restrict__ b2,
                               float* __restrict__ num_emb) {
    const int idx = blockIdx.x;        // [0, B*2C)
    const int b = idx >> 6, m = idx & 63;
    const int t = threadIdx.x;         // blockDim 128
    float x;
    if (m < CC) {
        const int hp = meta[4 * b], dbl = meta[4 * b + 1];
        const int pq0 = meta[4 * b + 2], pq1 = meta[4 * b + 3];
        x = hp ? conn[pq0 * CC + m] : 0.f;
        if (dbl) x += conn[pq1 * CC + m];
    } else {
        x = caps[b * CC + (m - CC)];
    }
    __shared__ float h[64];
    if (t < 64) { float v = W1[t] * x + b1[t]; h[t] = v > 0.f ? v : 0.f; }
    __syncthreads();
    float acc = b2[t];
    for (int j = 0; j < 64; ++j) acc += W2[t * 64 + j] * h[j];
    num_emb[(size_t)idx * EE + t] = acc > 0.f ? acc : 0.f;
}

// ---------------------------------------------------------------------------
// joined = aj2(relu(aj1([prev_oh, curr_oh]))) and pack core_info bf16 [B*C, 768]
// ---------------------------------------------------------------------------
__global__ void core_info_kernel(const int* __restrict__ prev,
                                 const int* __restrict__ curr,
                                 const int* __restrict__ meta,
                                 const float* __restrict__ a1W, const float* __restrict__ a1b,
                                 const float* __restrict__ a2W, const float* __restrict__ a2b,
                                 const float* __restrict__ num_emb,
                                 bfu* __restrict__ core_info) {
    const int idx = blockIdx.x;            // [0, B*C)
    const int b = idx >> 5, c = idx & 31;
    const int t = threadIdx.x;             // blockDim 256
    const int hp = meta[4 * b];
    const float a00 = a1W[0], a01 = a1W[1], a10 = a1W[2], a11 = a1W[3];
    const float ab0 = a1b[0], ab1 = a1b[1];
    const float w0 = a2W[0], w1 = a2W[1], wb = a2b[0];
    bfu* dst = core_info + (size_t)idx * 768;
    for (int q = t; q < QQ; q += 256) {
        float x0 = hp ? (prev[b * QQ + q] == c ? 1.f : 0.f) : 1.f;
        float x1 = (curr[b * QQ + q] == c) ? 1.f : 0.f;
        float y0 = a00 * x0 + a01 * x1 + ab0; y0 = y0 > 0.f ? y0 : 0.f;
        float y1 = a10 * x0 + a11 * x1 + ab1; y1 = y1 > 0.f ? y1 : 0.f;
        dst[q] = f2bf(w0 * y0 + w1 * y1 + wb);
    }
    dst[QQ + t] = f2bf(num_emb[(size_t)b * 8192 + c * 256 + t]);
}

// ---------------------------------------------------------------------------
// ctx = relu(cj3(relu(cj2(relu(cj1([circuit_emb, slice_adj]))))))  -> bf16
// ---------------------------------------------------------------------------
__global__ void ctx_kernel(const float* __restrict__ ce, const float* __restrict__ adj,
                           const float* __restrict__ c1W, const float* __restrict__ c1b,
                           const float* __restrict__ c2W, const float* __restrict__ c2b,
                           const float* __restrict__ c3W, const float* __restrict__ c3b,
                           bfu* __restrict__ ctx, long n) {
    const float a00 = c1W[0], a01 = c1W[1], a10 = c1W[2], a11 = c1W[3];
    const float ab0 = c1b[0], ab1 = c1b[1];
    const float d00 = c2W[0], d01 = c2W[1], d10 = c2W[2], d11 = c2W[3];
    const float db0 = c2b[0], db1 = c2b[1];
    const float w0 = c3W[0], w1 = c3W[1], wb = c3b[0];
    for (long i = blockIdx.x * (long)blockDim.x + threadIdx.x; i < n;
         i += (long)gridDim.x * blockDim.x) {
        float x0 = ce[i], x1 = adj[i];
        float y0 = a00 * x0 + a01 * x1 + ab0; y0 = y0 > 0.f ? y0 : 0.f;
        float y1 = a10 * x0 + a11 * x1 + ab1; y1 = y1 > 0.f ? y1 : 0.f;
        float z0 = d00 * y0 + d01 * y1 + db0; z0 = z0 > 0.f ? z0 : 0.f;
        float z1 = d10 * y0 + d11 * y1 + db1; z1 = z1 > 0.f ? z1 : 0.f;
        float v = w0 * z0 + w1 * z1 + wb;     v = v > 0.f ? v : 0.f;
        ctx[i] = f2bf(v);
    }
}

// ---------------------------------------------------------------------------
// Fused length-1-query MHA over ctx (keys = values = ctx[b], Lk = 512).
// One block per batch. Uses wk[h,:] = qh_h @ Wk_h so K/V are never projected.
// ---------------------------------------------------------------------------
__global__ __launch_bounds__(256)
void cmha_kernel(const float* __restrict__ qh,
                 const float* __restrict__ Wk, const float* __restrict__ bk,
                 const float* __restrict__ Wv, const float* __restrict__ bv,
                 const bfu* __restrict__ ctx, bfu* __restrict__ o_out) {
    const int b = blockIdx.x, t = threadIdx.x;  // blockDim 256
    __shared__ float qhL[QQ];
    __shared__ float wkL[HHD * QQ];   // later reused as obar
    __shared__ float sL[HHD * QQ];
    __shared__ float qkb[HHD];
    qhL[t] = qh[b * QQ + t];
    qhL[t + 256] = qh[b * QQ + t + 256];
    __syncthreads();
#pragma unroll 1
    for (int r = 0; r < 16; ++r) {                 // wk[h][i]
        const int j = t + 256 * r, h = j >> 9, i = j & 511;
        float a = 0.f;
        for (int d = 0; d < 64; ++d) a += qhL[h * 64 + d] * Wk[(h * 64 + d) * QQ + i];
        wkL[j] = a;
    }
    if (t < HHD) {
        float a = 0.f;
        for (int d = 0; d < 64; ++d) a += qhL[t * 64 + d] * bk[t * 64 + d];
        qkb[t] = a;
    }
    __syncthreads();
    const bfu* cb = ctx + (size_t)b * QQ * QQ;
#pragma unroll 1
    for (int r = 0; r < 16; ++r) {                 // scores
        const int h = t & 7, q = (t >> 3) + 32 * r;
        float a = qkb[h];
        const bfu* row = cb + (size_t)q * QQ;
        for (int i = 0; i < QQ; ++i) a += bf2f(row[i]) * wkL[h * QQ + i];
        sL[h * QQ + q] = a * 0.125f;
    }
    __syncthreads();
    {                                              // softmax: wave w handles head w
        const int h = t >> 5, lane = t & 31;
        float m = -1e30f;
        for (int q = lane; q < QQ; q += 32) m = fmaxf(m, sL[h * QQ + q]);
        for (int o = 16; o > 0; o >>= 1) m = fmaxf(m, __shfl_xor(m, o, 32));
        float s = 0.f;
        for (int q = lane; q < QQ; q += 32) {
            float e = __expf(sL[h * QQ + q] - m);
            sL[h * QQ + q] = e; s += e;
        }
        for (int o = 16; o > 0; o >>= 1) s += __shfl_xor(s, o, 32);
        const float inv = 1.f / s;
        for (int q = lane; q < QQ; q += 32) sL[h * QQ + q] *= inv;
    }
    __syncthreads();
    float a0[HHD], a1[HHD];                        // obar[h][i] = sum_q p * ctx
#pragma unroll
    for (int h = 0; h < HHD; ++h) { a0[h] = 0.f; a1[h] = 0.f; }
#pragma unroll 1
    for (int q = 0; q < QQ; ++q) {
        const float c0 = bf2f(cb[(size_t)q * QQ + t]);
        const float c1 = bf2f(cb[(size_t)q * QQ + t + 256]);
#pragma unroll
        for (int h = 0; h < HHD; ++h) {
            const float p = sL[h * QQ + q];
            a0[h] += p * c0; a1[h] += p * c1;
        }
    }
    __syncthreads();
#pragma unroll
    for (int h = 0; h < HHD; ++h) { wkL[h * QQ + t] = a0[h]; wkL[h * QQ + t + 256] = a1[h]; }
    __syncthreads();
#pragma unroll 1
    for (int r = 0; r < 2; ++r) {                  // o[j] = bv[j] + obar_h . Wv[j,:]
        const int j = t + 256 * r, h = j >> 6;
        float a = bv[j];
        for (int i = 0; i < QQ; ++i) a += wkL[h * QQ + i] * Wv[(size_t)j * QQ + i];
        o_out[(size_t)b * QQ + j] = f2bf(a);
    }
}

// ---------------------------------------------------------------------------
// Fused glimpse MHA (keys = values = core_embs[b], Lk = 32). Also emits
// attn_w = mean over heads -> d_out[0:4096]. K/V tile staged in LDS.
// ---------------------------------------------------------------------------
__global__ __launch_bounds__(256)
void glimpse_kernel(const float* __restrict__ qh2,
                    const float* __restrict__ Wk, const float* __restrict__ bk,
                    const float* __restrict__ Wv, const float* __restrict__ bv,
                    const bfu* __restrict__ ce,
                    float* __restrict__ attn_out, bfu* __restrict__ gpre) {
    const int b = blockIdx.x, t = threadIdx.x;  // blockDim 256
    __shared__ float qhL[QQ];
    __shared__ bfu   ceL[CC * QQ];
    __shared__ float wkL[HHD * QQ];   // reused as obar
    __shared__ float sL[HHD * CC];
    __shared__ float qkb[HHD];
    qhL[t] = qh2[b * QQ + t];
    qhL[t + 256] = qh2[b * QQ + t + 256];
    const bfu* src = ce + (size_t)b * CC * QQ;
#pragma unroll 1
    for (int r = 0; r < 64; ++r) ceL[t + 256 * r] = src[t + 256 * r];
    __syncthreads();
#pragma unroll 1
    for (int r = 0; r < 16; ++r) {
        const int j = t + 256 * r, h = j >> 9, i = j & 511;
        float a = 0.f;
        for (int d = 0; d < 64; ++d) a += qhL[h * 64 + d] * Wk[(h * 64 + d) * QQ + i];
        wkL[j] = a;
    }
    if (t < HHD) {
        float a = 0.f;
        for (int d = 0; d < 64; ++d) a += qhL[t * 64 + d] * bk[t * 64 + d];
        qkb[t] = a;
    }
    __syncthreads();
    {                                              // scores: 8 heads x 32 cores
        const int h = t >> 5, c = t & 31;
        float a = qkb[h];
        for (int i = 0; i < QQ; ++i) a += bf2f(ceL[c * QQ + i]) * wkL[h * QQ + i];
        sL[h * CC + c] = a * 0.125f;
    }
    __syncthreads();
    {                                              // softmax within each wave (32 = C)
        const int h = t >> 5, c = t & 31;
        float v = sL[h * CC + c];
        float m = v;
        for (int o = 16; o > 0; o >>= 1) m = fmaxf(m, __shfl_xor(m, o, 32));
        float e = __expf(v - m);
        float s = e;
        for (int o = 16; o > 0; o >>= 1) s += __shfl_xor(s, o, 32);
        sL[h * CC + c] = e / s;
    }
    __syncthreads();
    if (t < CC) {                                  // attn_w output: head mean
        float a = 0.f;
        for (int h = 0; h < HHD; ++h) a += sL[h * CC + t];
        attn_out[b * CC + t] = a * 0.125f;
    }
    float a0[HHD], a1[HHD];
#pragma unroll
    for (int h = 0; h < HHD; ++h) { a0[h] = 0.f; a1[h] = 0.f; }
#pragma unroll 1
    for (int c = 0; c < CC; ++c) {
        const float c0 = bf2f(ceL[c * QQ + t]);
        const float c1 = bf2f(ceL[c * QQ + t + 256]);
#pragma unroll
        for (int h = 0; h < HHD; ++h) {
            const float p = sL[h * CC + c];
            a0[h] += p * c0; a1[h] += p * c1;
        }
    }
    __syncthreads();
#pragma unroll
    for (int h = 0; h < HHD; ++h) { wkL[h * QQ + t] = a0[h]; wkL[h * QQ + t + 256] = a1[h]; }
    __syncthreads();
#pragma unroll 1
    for (int r = 0; r < 2; ++r) {
        const int j = t + 256 * r, h = j >> 6;
        float a = bv[j];
        for (int i = 0; i < QQ; ++i) a += wkL[h * QQ + i] * Wv[(size_t)j * QQ + i];
        gpre[(size_t)b * QQ + j] = f2bf(a);
    }
}

// ---------------------------------------------------------------------------
__global__ void value_kernel(const bfu* __restrict__ h3, const float* __restrict__ W,
                             const float* __restrict__ bsc, float* __restrict__ out) {
    const int b = blockIdx.x, t = threadIdx.x;  // blockDim 64
    __shared__ float red[64];
    float a = 0.f;
    for (int i = t; i < QQ; i += 64) a += bf2f(h3[(size_t)b * QQ + i]) * W[i];
    red[t] = a;
    __syncthreads();
    for (int s = 32; s > 0; s >>= 1) { if (t < s) red[t] += red[t + s]; __syncthreads(); }
    if (t == 0) out[b] = red[0] + bsc[0];
}

// ---------------------------------------------------------------------------
extern "C" void kernel_launch(void* const* d_in, const int* in_sizes, int n_in,
                              void* d_out, int out_size, void* d_ws, size_t ws_size,
                              hipStream_t stream) {
    (void)in_sizes; (void)n_in; (void)out_size; (void)ws_size;
    // inputs
    const int*   qubits = (const int*)d_in[0];
    const int*   prev   = (const int*)d_in[1];
    const int*   curr   = (const int*)d_in[2];
    const float* caps   = (const float*)d_in[3];
    const float* conn   = (const float*)d_in[4];
    const float* cemb   = (const float*)d_in[5];
    const float* sadj   = (const float*)d_in[6];
    // params (flattened in insertion order; each Linear = (W, b), MHA = (Wqkv,bqkv,Wo,bo))
    const float* num1W = (const float*)d_in[7];  const float* num1b = (const float*)d_in[8];
    const float* num2W = (const float*)d_in[9];  const float* num2b = (const float*)d_in[10];
    const float* aj1W  = (const float*)d_in[11]; const float* aj1b  = (const float*)d_in[12];
    const float* aj2W  = (const float*)d_in[13]; const float* aj2b  = (const float*)d_in[14];
    const float* ce1W  = (const float*)d_in[15]; const float* ce1b  = (const float*)d_in[16];
    const float* ce2W  = (const float*)d_in[17]; const float* ce2b  = (const float*)d_in[18];
    const float* cj1W  = (const float*)d_in[19]; const float* cj1b  = (const float*)d_in[20];
    const float* cj2W  = (const float*)d_in[21]; const float* cj2b  = (const float*)d_in[22];
    const float* cj3W  = (const float*)d_in[23]; const float* cj3b  = (const float*)d_in[24];
    const float* cWqkv = (const float*)d_in[25]; const float* cbqkv = (const float*)d_in[26];
    const float* cWo   = (const float*)d_in[27]; const float* cbo   = (const float*)d_in[28];
    const float* cf1W  = (const float*)d_in[29]; const float* cf1b  = (const float*)d_in[30];
    const float* cf2W  = (const float*)d_in[31]; const float* cf2b  = (const float*)d_in[32];
    const float* mWqkv = (const float*)d_in[33]; const float* mbqkv = (const float*)d_in[34];
    const float* mWo   = (const float*)d_in[35]; const float* mbo   = (const float*)d_in[36];
    const float* v1W   = (const float*)d_in[37]; const float* v1b   = (const float*)d_in[38];
    const float* v2W   = (const float*)d_in[39]; const float* v2b   = (const float*)d_in[40];
    const float* v3W   = (const float*)d_in[41]; const float* v3b   = (const float*)d_in[42];
    const float* v4W   = (const float*)d_in[43]; const float* v4b   = (const float*)d_in[44];

    // workspace bump allocator
    char* wsp = (char*)d_ws;
    auto alloc = [&](size_t bytes) -> void* {
        void* p = (void*)wsp;
        wsp += (bytes + 255) & ~(size_t)255;
        return p;
    };
    int*   meta      = (int*)  alloc(BB * 4 * sizeof(int));
    bfu*   qm        = (bfu*)  alloc((size_t)BB * QQ * 2);
    float* num_emb   = (float*)alloc((size_t)BB * 64 * EE * 4);
    bfu*   core_info = (bfu*)  alloc((size_t)BB * CC * 768 * 2);
    bfu*   h1        = (bfu*)  alloc((size_t)BB * CC * QQ * 2);
    bfu*   core_embs = (bfu*)  alloc((size_t)BB * CC * QQ * 2);
    bfu*   ctx       = (bfu*)  alloc((size_t)BB * QQ * QQ * 2);
    float* qh_c      = (float*)alloc((size_t)BB * QQ * 4);
    bfu*   o_mha     = (bfu*)  alloc((size_t)BB * QQ * 2);
    bfu*   actx0     = (bfu*)  alloc((size_t)BB * QQ * 2);
    bfu*   a1buf     = (bfu*)  alloc((size_t)BB * QQ * 2);
    bfu*   v_in      = (bfu*)  alloc((size_t)BB * 1024 * 2);
    float* qh2       = (float*)alloc((size_t)BB * QQ * 4);
    bfu*   gpre      = (bfu*)  alloc((size_t)BB * QQ * 2);
    bfu*   hv1       = (bfu*)  alloc((size_t)BB * QQ * 2);
    bfu*   hv2       = (bfu*)  alloc((size_t)BB * QQ * 2);
    bfu*   hv3       = (bfu*)  alloc((size_t)BB * QQ * 2);
    // bf16 weight copies for WMMA (layout [N, K] row-major, as stored)
    bfu* ce1Wb = (bfu*)alloc((size_t)512 * 768 * 2);
    bfu* ce2Wb = (bfu*)alloc((size_t)512 * 512 * 2);
    bfu* WqCb  = (bfu*)alloc((size_t)512 * 512 * 2);
    bfu* WoCb  = (bfu*)alloc((size_t)512 * 512 * 2);
    bfu* cf1Wb = (bfu*)alloc((size_t)512 * 512 * 2);
    bfu* cf2Wb = (bfu*)alloc((size_t)512 * 512 * 2);
    bfu* Wq2b  = (bfu*)alloc((size_t)512 * 512 * 2);
    bfu* Wo2b  = (bfu*)alloc((size_t)512 * 512 * 2);
    bfu* v1Wb  = (bfu*)alloc((size_t)512 * 1024 * 2);
    bfu* v2Wb  = (bfu*)alloc((size_t)512 * 512 * 2);
    bfu* v3Wb  = (bfu*)alloc((size_t)512 * 512 * 2);

    // weight conversions
    cvt_f32_bf16<<<512, 256, 0, stream>>>(ce1W, ce1Wb, 512 * 768);
    cvt_f32_bf16<<<512, 256, 0, stream>>>(ce2W, ce2Wb, 512 * 512);
    cvt_f32_bf16<<<512, 256, 0, stream>>>(cWqkv, WqCb, 512 * 512);   // Wq of cmha
    cvt_f32_bf16<<<512, 256, 0, stream>>>(cWo, WoCb, 512 * 512);
    cvt_f32_bf16<<<512, 256, 0, stream>>>(cf1W, cf1Wb, 512 * 512);
    cvt_f32_bf16<<<512, 256, 0, stream>>>(cf2W, cf2Wb, 512 * 512);
    cvt_f32_bf16<<<512, 256, 0, stream>>>(mWqkv, Wq2b, 512 * 512);   // Wq of mha
    cvt_f32_bf16<<<512, 256, 0, stream>>>(mWo, Wo2b, 512 * 512);
    cvt_f32_bf16<<<512, 256, 0, stream>>>(v1W, v1Wb, 512 * 1024);
    cvt_f32_bf16<<<512, 256, 0, stream>>>(v2W, v2Wb, 512 * 512);
    cvt_f32_bf16<<<512, 256, 0, stream>>>(v3W, v3Wb, 512 * 512);

    // prep
    batch_prep<<<BB, 128, 0, stream>>>(qubits, prev, meta, qm);
    num_emb_kernel<<<BB * 64, 128, 0, stream>>>(conn, caps, meta, num1W, num1b,
                                                num2W, num2b, num_emb);
    core_info_kernel<<<BB * CC, 256, 0, stream>>>(prev, curr, meta, aj1W, aj1b,
                                                  aj2W, aj2b, num_emb, core_info);
    ctx_kernel<<<4096, 256, 0, stream>>>(cemb, sadj, cj1W, cj1b, cj2W, cj2b,
                                         cj3W, cj3b, ctx, (long)BB * QQ * QQ);

    // core embeddings: ce1 + relu, ce2 + relu
    gemm_bf16_wmma<<<dim3(32, 32), 256, 0, stream>>>(core_info, 768, ce1Wb, ce1b,
                                                     768, 1, h1, 512, nullptr, 0);
    gemm_bf16_wmma<<<dim3(32, 32), 256, 0, stream>>>(h1, 512, ce2Wb, ce2b,
                                                     512, 1, core_embs, 512, nullptr, 0);

    // cmha: q projection (f32 out), fused attention, out-proj, cf1, cf2
    gemm_bf16_wmma<<<dim3(1, 32), 256, 0, stream>>>(qm, 512, WqCb, cbqkv,
                                                    512, 0, nullptr, 0, qh_c, 512);
    cmha_kernel<<<BB, 256, 0, stream>>>(qh_c, cWqkv + 512 * 512, cbqkv + 512,
                                        cWqkv + 2 * 512 * 512, cbqkv + 1024, ctx, o_mha);
    gemm_bf16_wmma<<<dim3(1, 32), 256, 0, stream>>>(o_mha, 512, WoCb, cbo,
                                                    512, 0, actx0, 512, nullptr, 0);
    gemm_bf16_wmma<<<dim3(1, 32), 256, 0, stream>>>(actx0, 512, cf1Wb, cf1b,
                                                    512, 1, a1buf, 512, nullptr, 0);
    gemm_bf16_wmma<<<dim3(1, 32), 256, 0, stream>>>(a1buf, 512, cf2Wb, cf2b,
                                                    512, 1, v_in + 512, 1024, nullptr, 0);

    // glimpse mha (attn_w -> d_out), out-proj into v_in[:, 0:512]
    gemm_bf16_wmma<<<dim3(1, 32), 256, 0, stream>>>(v_in + 512, 1024, Wq2b, mbqkv,
                                                    512, 0, nullptr, 0, qh2, 512);
    glimpse_kernel<<<BB, 256, 0, stream>>>(qh2, mWqkv + 512 * 512, mbqkv + 512,
                                           mWqkv + 2 * 512 * 512, mbqkv + 1024,
                                           core_embs, (float*)d_out, gpre);
    gemm_bf16_wmma<<<dim3(1, 32), 256, 0, stream>>>(gpre, 512, Wo2b, mbo,
                                                    512, 0, v_in, 1024, nullptr, 0);

    // value head
    gemm_bf16_wmma<<<dim3(1, 32), 256, 0, stream>>>(v_in, 1024, v1Wb, v1b,
                                                    1024, 1, hv1, 512, nullptr, 0);
    gemm_bf16_wmma<<<dim3(1, 32), 256, 0, stream>>>(hv1, 512, v2Wb, v2b,
                                                    512, 1, hv2, 512, nullptr, 0);
    gemm_bf16_wmma<<<dim3(1, 32), 256, 0, stream>>>(hv2, 512, v3Wb, v3b,
                                                    512, 1, hv3, 512, nullptr, 0);
    value_kernel<<<BB, 64, 0, stream>>>(hv3, v4W, v4b, (float*)d_out + BB * CC);
}
